// DecoderLayer_23218593202573
// MI455X (gfx1250) — matrix operand order repaired and checked by reading, not compiled, not run on hardware
//
#include <hip/hip_runtime.h>
#include <hip/hip_bf16.h>
#include <math.h>
#include <stdint.h>

typedef __attribute__((ext_vector_type(16))) _Float16 v16h;
typedef __attribute__((ext_vector_type(8)))  float    v8f;

__device__ __forceinline__ v8f wmma16x16x32(v16h a, v16h b, v8f c) {
  return __builtin_amdgcn_wmma_f32_16x16x32_f16(false, a, false, b, (short)0, c, false, false);
}

__device__ __forceinline__ float gelu_exact(float x) {
  return 0.5f * x * (1.0f + erff(x * 0.70710678118654752f));
}

__device__ __forceinline__ uint32_t lds_addr(const void* p) {
  // Flat LDS addresses carry the workgroup-relative offset in the low 32 bits.
  return (uint32_t)(unsigned long long)p;
}

__device__ __forceinline__ void async_b128(uint32_t lds, uint32_t voff, uint64_t sbase) {
  asm volatile("global_load_async_to_lds_b128 %0, %1, %2"
               :: "v"(lds), "v"(voff), "s"(sbase) : "memory");
}
__device__ __forceinline__ void wait_async0() {
  asm volatile("s_wait_asynccnt 0" ::: "memory");
}
__device__ __forceinline__ void wait_async8() {
  // Leave the 8 just-issued prefetch loads outstanding; in-order completion
  // means the previous tile's 8 loads have landed.
  asm volatile("s_wait_asynccnt 8" ::: "memory");
}

// ---------------------------------------------------------------------------
// LayerNorm (Bessel ddof=1, eps on std). Writes fp32 (residual) + f16 (GEMM A).
// ---------------------------------------------------------------------------
__global__ __launch_bounds__(256) void ln_kernel(const float* __restrict__ in,
                                                 const float* __restrict__ g,
                                                 const float* __restrict__ bb,
                                                 float* __restrict__ out32,
                                                 _Float16* __restrict__ out16) {
  __shared__ float red[16];
  const int row = blockIdx.x;
  const int tid = threadIdx.x;
  const float* p = in + (size_t)row * 1024;
  float s = 0.f, s2 = 0.f;
#pragma unroll
  for (int i = 0; i < 4; ++i) { float v = p[tid + i * 256]; s += v; s2 += v * v; }
#pragma unroll
  for (int off = 16; off; off >>= 1) { s += __shfl_xor(s, off); s2 += __shfl_xor(s2, off); }
  const int lane = tid & 31, wave = tid >> 5;
  if (lane == 0) { red[wave] = s; red[8 + wave] = s2; }
  __syncthreads();
  if (wave == 0) {
    s  = (lane < 8) ? red[lane] : 0.f;
    s2 = (lane < 8) ? red[8 + lane] : 0.f;
#pragma unroll
    for (int off = 4; off; off >>= 1) { s += __shfl_xor(s, off); s2 += __shfl_xor(s2, off); }
    if (lane == 0) { red[0] = s; red[1] = s2; }
  }
  __syncthreads();
  s = red[0]; s2 = red[1];
  const float mean = s * (1.0f / 1024.0f);
  const float var  = (s2 - 1024.0f * mean * mean) * (1.0f / 1023.0f);
  const float inv  = 1.0f / (sqrtf(fmaxf(var, 0.f)) + 1e-6f);
#pragma unroll
  for (int i = 0; i < 4; ++i) {
    int c = tid + i * 256;
    float v = g[c] * (p[c] - mean) * inv + bb[c];
    out32[(size_t)row * 1024 + c] = v;
    out16[(size_t)row * 1024 + c] = (_Float16)v;
  }
}

// ---------------------------------------------------------------------------
// Weight convert + transpose: dst[n*K + k] = (f16)src[k*N + n]
// ---------------------------------------------------------------------------
__global__ __launch_bounds__(256) void cvt_t_kernel(const float* __restrict__ src,
                                                    _Float16* __restrict__ dst,
                                                    int K, int N) {
  __shared__ float t[32][33];
  const int bn = blockIdx.x * 32, bk = blockIdx.y * 32;
  const int tx = threadIdx.x & 31, ty = threadIdx.x >> 5;  // 32 x 8
#pragma unroll
  for (int i = 0; i < 32; i += 8)
    t[ty + i][tx] = src[(size_t)(bk + ty + i) * N + bn + tx];
  __syncthreads();
#pragma unroll
  for (int i = 0; i < 32; i += 8)
    dst[(size_t)(bn + ty + i) * K + bk + tx] = (_Float16)t[tx][ty + i];
}

// ---------------------------------------------------------------------------
// WMMA GEMM, all-f16 operands: C = epi(A[M][K] @ Bt[N][K]^T + bias)
// Block tile 128x128, BK=64, 256 threads = 8 waves (4m x 2n), wave = 32x64
// (2x4 WMMA tiles, 16 v_wmma per stage). Double-buffered async-to-LDS
// staging, software-pipelined with s_wait_asynccnt 8 partial waits.
// ---------------------------------------------------------------------------
#define BM 128
#define BN 128
#define BK 64
#define LDT 72  // halves per LDS row (144B, 16B-aligned)

__device__ __forceinline__ void gemm_stage(uint64_t abase, uint64_t bbase,
                                           uint32_t sAb, uint32_t sBb,
                                           int tid, int K) {
#pragma unroll
  for (int j = 0; j < 4; ++j) {
    int c = tid + j * 256;
    int row = c >> 3, piece = c & 7;
    uint32_t voff = (uint32_t)(row * K + piece * 8) * 2;
    uint32_t loff = (uint32_t)(row * LDT + piece * 8) * 2;
    async_b128(sAb + loff, voff, abase);
    async_b128(sBb + loff, voff, bbase);
  }
}

template <int ACT, bool RES, bool OUT32, bool OUT16>
__global__ __launch_bounds__(256) void gemm_wmma_kernel(
    const _Float16* __restrict__ A, const _Float16* __restrict__ Bt,
    const float* __restrict__ bias, const float* __restrict__ residual,
    float* __restrict__ C, _Float16* __restrict__ C16, int N, int K) {
  __shared__ _Float16 sA[2][BM * LDT];
  __shared__ _Float16 sB[2][BN * LDT];

  const int tid  = threadIdx.x;
  const int lane = tid & 31;
  const int wave = tid >> 5;
  const int wm   = wave >> 1;   // 0..3 (32 rows)
  const int wn   = wave & 1;    // 0..1 (64 cols)
  const int lg   = lane >> 4;
  const int ln16 = lane & 15;

  const uint64_t Ab = (uint64_t)(unsigned long long)(A  + (size_t)blockIdx.y * BM * K);
  const uint64_t Bb = (uint64_t)(unsigned long long)(Bt + (size_t)blockIdx.x * BN * K);
  const uint32_t sAd[2] = { lds_addr(sA[0]), lds_addr(sA[1]) };
  const uint32_t sBd[2] = { lds_addr(sB[0]), lds_addr(sB[1]) };

  v8f acc[2][4] = {};

  const int ntile = K / BK;
  gemm_stage(Ab, Bb, sAd[0], sBd[0], tid, K);  // tile 0 -> buf 0

  for (int i = 0; i < ntile; ++i) {
    const int cur = i & 1;
    if (i + 1 < ntile) {
      const uint64_t ko = (uint64_t)((i + 1) * BK * 2);
      gemm_stage(Ab + ko, Bb + ko, sAd[cur ^ 1], sBd[cur ^ 1], tid, K);
      wait_async8();  // tile i landed; tile i+1 still in flight
    } else {
      wait_async0();
    }
    __syncthreads();

    const _Float16* __restrict__ cA = sA[cur];
    const _Float16* __restrict__ cB = sB[cur];
#pragma unroll
    for (int kc = 0; kc < 2; ++kc) {
      v16h af[2], bf[4];
      const int kb = kc * 32 + lg * 16;
#pragma unroll
      for (int tm = 0; tm < 2; ++tm) {
        const _Float16* p = &cA[(wm * 32 + tm * 16 + ln16) * LDT + kb];
#pragma unroll
        for (int h = 0; h < 16; ++h) af[tm][h] = p[h];
      }
      const int kb2 = kc * 32 + lg * 16;
#pragma unroll
      for (int tn = 0; tn < 4; ++tn) {
        const _Float16* q = &cB[(wn * 64 + tn * 16 + ln16) * LDT + kb2];
#pragma unroll
        for (int h = 0; h < 16; ++h) bf[tn][h] = q[h];
      }
#pragma unroll
      for (int tm = 0; tm < 2; ++tm)
#pragma unroll
        for (int tn = 0; tn < 4; ++tn)
          acc[tm][tn] = wmma16x16x32(af[tm], bf[tn], acc[tm][tn]);
    }
    __syncthreads();
  }

  // ---- straight-line epilogue (templated) ----
#pragma unroll
  for (int tm = 0; tm < 2; ++tm) {
#pragma unroll
    for (int tn = 0; tn < 4; ++tn) {
      const int rbase = blockIdx.y * BM + wm * 32 + tm * 16 + lg * 8;
      const int col   = blockIdx.x * BN + wn * 64 + tn * 16 + ln16;
      const float bv  = bias[col];
#pragma unroll
      for (int r = 0; r < 8; ++r) {
        const size_t o = (size_t)(rbase + r) * N + col;
        float v = acc[tm][tn][r] + bv;
        if (ACT == 1) v = gelu_exact(v);
        if (RES) v += residual[o];
        if (OUT32) C[o] = v;
        if (OUT16) C16[o] = (_Float16)v;
      }
    }
  }
}

// ---------------------------------------------------------------------------
// Causal flash attention on f16 qkv [4096][3072] (q|k|v, head h at cols h*64).
// Block = (b, h, 64 q rows); 128 threads = 4 waves x 16 q rows. K tile async
// to LDS [key][d]; V tile transposed to [d][key] so P@V B-frags are b128.
// ---------------------------------------------------------------------------
#define LKV 72  // sK row stride (halves)
#define LVT 40  // sVt/sP row stride (halves)

__global__ __launch_bounds__(128) void attn_kernel(const _Float16* __restrict__ qkv,
                                                   _Float16* __restrict__ out) {
  __shared__ _Float16 sK[32 * LKV];
  __shared__ _Float16 sVt[64 * LVT];
  __shared__ _Float16 sP[4][16 * LVT];

  const int tid  = threadIdx.x;
  const int lane = tid & 31;
  const int wave = tid >> 5;
  const int lg   = lane >> 4;
  const int ln16 = lane & 15;

  const int qb = blockIdx.x & 31;
  const int hh = (blockIdx.x >> 5) & 15;
  const int b  = blockIdx.x >> 9;
  const int q0 = qb * 64 + wave * 16;

  // Q fragments (two 32-wide chunks of head_dim=64), softmax scale folded in.
  v16h qf[2];
  {
    const _Float16* qptr = qkv + (size_t)(b * 2048 + q0 + ln16) * 3072 + hh * 64;
    const int kb = lg * 16;
#pragma unroll
    for (int c = 0; c < 2; ++c)
#pragma unroll
      for (int h = 0; h < 16; ++h) {
        int d = c * 32 + kb + h;
        qf[c][h] = qptr[d] * (_Float16)0.125f;
      }
  }

  v8f acc[4] = {};
  float mrow[8], lrow[8];
#pragma unroll
  for (int r = 0; r < 8; ++r) { mrow[r] = -1e30f; lrow[r] = 0.f; }

  const uint64_t kbase = (uint64_t)(unsigned long long)qkv +
                         (uint64_t)((size_t)b * 2048 * 3072 + 1024 + hh * 64) * 2;
  const uint32_t sK0 = lds_addr(sK);

  const int nkt = 2 * qb + 2;
  for (int kt = 0; kt < nkt; ++kt) {
    // ---- K tile: 32 keys x 128B, async (256 chunks / 128 threads) ----
#pragma unroll
    for (int j = 0; j < 2; ++j) {
      int c = tid + j * 128;
      int key = c >> 3, piece = c & 7;
      uint32_t voff = (uint32_t)((kt * 32 + key) * 6144 + piece * 16);
      uint32_t loff = (uint32_t)(key * LKV + piece * 8) * 2;
      async_b128(sK0 + loff, voff, kbase);
    }
    // ---- V tile: load b128, scatter-transpose to sVt[d][key] ----
#pragma unroll
    for (int j = 0; j < 2; ++j) {
      int c = tid + j * 128;
      int key = c >> 3, piece = c & 7;
      const _Float16* vp = qkv + (size_t)(b * 2048 + kt * 32 + key) * 3072 +
                           2048 + hh * 64 + piece * 8;
      union { uint4 u; _Float16 h[8]; } raw;
      raw.u = *(const uint4*)vp;
#pragma unroll
      for (int e = 0; e < 8; ++e) sVt[(piece * 8 + e) * LVT + key] = raw.h[e];
    }
    wait_async0();
    __syncthreads();

    // ---- scores: S[16q x 32k] = Q @ K^T (4 WMMAs) ----
    v8f sfr[2] = {};
#pragma unroll
    for (int t = 0; t < 2; ++t) {
      const _Float16* kp = &sK[(t * 16 + ln16) * LKV + lg * 16];
#pragma unroll
      for (int c = 0; c < 2; ++c) {
        v16h bfv;
#pragma unroll
        for (int h = 0; h < 16; ++h) bfv[h] = kp[c * 32 + h];
        sfr[t] = wmma16x16x32(qf[c], bfv, sfr[t]);
      }
    }

    // ---- causal mask + online softmax ----
    const int myrow = q0 + lg * 8;
    const int col0  = kt * 32 + ln16;
#pragma unroll
    for (int t = 0; t < 2; ++t) {
      const int col = col0 + t * 16;
#pragma unroll
      for (int r = 0; r < 8; ++r)
        if (col > myrow + r) sfr[t][r] = -1e30f;
    }
#pragma unroll
    for (int r = 0; r < 8; ++r) {
      float m = fmaxf(sfr[0][r], sfr[1][r]);
      m = fmaxf(m, __shfl_xor(m, 1));
      m = fmaxf(m, __shfl_xor(m, 2));
      m = fmaxf(m, __shfl_xor(m, 4));
      m = fmaxf(m, __shfl_xor(m, 8));
      const float mnew = fmaxf(mrow[r], m);
      const float sc   = expf(mrow[r] - mnew);
      mrow[r] = mnew;
      float p0 = expf(sfr[0][r] - mnew);
      float p1 = expf(sfr[1][r] - mnew);
      sfr[0][r] = p0; sfr[1][r] = p1;
      float ps = p0 + p1;
      ps += __shfl_xor(ps, 1);
      ps += __shfl_xor(ps, 2);
      ps += __shfl_xor(ps, 4);
      ps += __shfl_xor(ps, 8);
      lrow[r] = lrow[r] * sc + ps;
      acc[0][r] *= sc; acc[1][r] *= sc; acc[2][r] *= sc; acc[3][r] *= sc;
    }

    // ---- transpose P through per-wave LDS into A-fragment layout ----
#pragma unroll
    for (int t = 0; t < 2; ++t)
#pragma unroll
      for (int r = 0; r < 8; ++r)
        sP[wave][(r + lg * 8) * LVT + t * 16 + ln16] = (_Float16)sfr[t][r];
    __syncthreads();

    v16h pf;
    {
      const _Float16* p = &sP[wave][ln16 * LVT + lg * 16];
#pragma unroll
      for (int h = 0; h < 16; ++h) pf[h] = p[h];
    }
    // ---- acc += P(16x32) @ V(32x64): B-frags contiguous from sVt ----
#pragma unroll
    for (int dt = 0; dt < 4; ++dt) {
      v16h vf;
      const _Float16* vq = &sVt[(dt * 16 + ln16) * LVT + lg * 16];
#pragma unroll
      for (int h = 0; h < 16; ++h) vf[h] = vq[h];
      acc[dt] = wmma16x16x32(pf, vf, acc[dt]);
    }
    __syncthreads();
  }

  const int orow = q0 + lg * 8;
#pragma unroll
  for (int dt = 0; dt < 4; ++dt)
#pragma unroll
    for (int r = 0; r < 8; ++r)
      out[(size_t)(b * 2048 + orow + r) * 1024 + hh * 64 + dt * 16 + ln16] =
          (_Float16)(acc[dt][r] / lrow[r]);
}

// ---------------------------------------------------------------------------
// Launcher
// ---------------------------------------------------------------------------
extern "C" void kernel_launch(void* const* d_in, const int* in_sizes, int n_in,
                              void* d_out, int out_size, void* d_ws, size_t ws_size,
                              hipStream_t stream) {
  const float* x     = (const float*)d_in[0];
  const float* w_qkv = (const float*)d_in[1];
  const float* b_qkv = (const float*)d_in[2];
  const float* w_out = (const float*)d_in[3];
  const float* b_out = (const float*)d_in[4];
  const float* ln1_a = (const float*)d_in[5];
  const float* ln1_b = (const float*)d_in[6];
  const float* ln2_a = (const float*)d_in[7];
  const float* ln2_b = (const float*)d_in[8];
  const float* w1    = (const float*)d_in[9];
  const float* b1    = (const float*)d_in[10];
  const float* w2    = (const float*)d_in[11];
  const float* b2    = (const float*)d_in[12];
  float* out = (float*)d_out;

  const int T = 4096, D = 1024, DM = 4096;
  unsigned char* w = (unsigned char*)d_ws;
  float* xn32 = (float*)w;      w += (size_t)T * D * 4;
  float* z    = (float*)w;      w += (size_t)T * D * 4;
  float* zn32 = (float*)w;      w += (size_t)T * D * 4;
  _Float16* xn16   = (_Float16*)w; w += (size_t)T * D * 2;
  _Float16* zn16   = (_Float16*)w; w += (size_t)T * D * 2;
  _Float16* qkv16  = (_Float16*)w; w += (size_t)T * 3 * D * 2;
  _Float16* attn16 = (_Float16*)w; w += (size_t)T * D * 2;
  _Float16* h16    = (_Float16*)w; w += (size_t)T * DM * 2;
  _Float16* wqkvT  = (_Float16*)w; w += (size_t)D * 3 * D * 2;
  _Float16* woutT  = (_Float16*)w; w += (size_t)D * D * 2;
  _Float16* w1T    = (_Float16*)w; w += (size_t)D * DM * 2;
  _Float16* w2T    = (_Float16*)w; w += (size_t)DM * D * 2;

  // 0) convert + transpose weights to f16 [N][K]
  cvt_t_kernel<<<dim3(3 * D / 32, D / 32), 256, 0, stream>>>(w_qkv, wqkvT, D, 3 * D);
  cvt_t_kernel<<<dim3(D / 32, D / 32),     256, 0, stream>>>(w_out, woutT, D, D);
  cvt_t_kernel<<<dim3(DM / 32, D / 32),    256, 0, stream>>>(w1, w1T, D, DM);
  cvt_t_kernel<<<dim3(D / 32, DM / 32),    256, 0, stream>>>(w2, w2T, DM, D);

  // 1) xn = LN1(x)
  ln_kernel<<<T, 256, 0, stream>>>(x, ln1_a, ln1_b, xn32, xn16);
  // 2) qkv16 = f16(xn @ w_qkv + b_qkv)
  gemm_wmma_kernel<0, false, false, true><<<dim3(3 * D / BN, T / BM), 256, 0, stream>>>(
      xn16, wqkvT, b_qkv, nullptr, nullptr, qkv16, 3 * D, D);
  // 3) attn16 = causal softmax(q k^T / 8) v
  attn_kernel<<<2 * 16 * 32, 128, 0, stream>>>(qkv16, attn16);
  // 4) z = xn + attn @ w_out + b_out
  gemm_wmma_kernel<0, true, true, false><<<dim3(D / BN, T / BM), 256, 0, stream>>>(
      attn16, woutT, b_out, xn32, z, nullptr, D, D);
  // 5) zn = LN2(z)
  ln_kernel<<<T, 256, 0, stream>>>(z, ln2_a, ln2_b, zn32, zn16);
  // 6) h16 = f16(gelu(zn @ w1 + b1))
  gemm_wmma_kernel<1, false, false, true><<<dim3(DM / BN, T / BM), 256, 0, stream>>>(
      zn16, w1T, b1, nullptr, nullptr, h16, DM, D);
  // 7) out = zn + h @ w2 + b2
  gemm_wmma_kernel<0, true, true, false><<<dim3(D / BN, T / BM), 256, 0, stream>>>(
      h16, w2T, b2, zn32, out, nullptr, D, DM);
}
